// ST_GRL_Model_77601469104344
// MI455X (gfx1250) — compile-verified, matrix operand order checked by reading, not run on hardware
//
#include <hip/hip_runtime.h>
#include <hip/hip_bf16.h>

// ============================================================================
// ST-GRL forward for gfx1250 (MI455X). All large GEMMs go through
// v_wmma_f32_16x16x32_f16 with fp32 hi/lo operand splitting (3 WMMAs / k-step).
// Weights are pre-packed into WMMA B-fragment order (zero-padded) so the GEMM
// inner loop is pure b128 loads + cvt + wmma. Epilogue (bias / exact GELU /
// residual) is a compile-time template specialization -> branch-free stores.
// Model constants: N=32, CIN=3, T=256, V=25, D=256, DIN=128, H=4, W=20, NC=60.
// ============================================================================

typedef __attribute__((ext_vector_type(16))) _Float16 v16h;
typedef __attribute__((ext_vector_type(8)))  float    v8f;

#define GF_GELU 1
#define GF_RES  2

__device__ __forceinline__ float gelu_exact(float x) {
  return 0.5f * x * (1.0f + erff(x * 0.70710678118654752440f));
}

// ---------------------------------------------------------------------------
// Pack W[K,N] (row-major) into WMMA B-fragment order, zero-padded to nT16*16
// columns. Layout: Bp[((kchunk*nT16 + ntile)*32 + lane)*16 + e], where
//   n     = ntile*16 + (lane&15)
//   khalf = (lane>>4)*8
//   k     = kchunk*32 + (e<8 ? khalf+e : 16+khalf+(e-8))
// ---------------------------------------------------------------------------
__global__ void k_packB(const float* __restrict__ W, float* __restrict__ Bp,
                        int K, int N, int nT16)
{
  int id = blockIdx.x * 256 + threadIdx.x;
  int total = (K >> 5) * nT16 * 512;
  if (id >= total) return;
  int e    = id & 15;
  int lane = (id >> 4) & 31;
  int nt   = (id >> 9) % nT16;
  int c    = id / (512 * nT16);
  int n    = nt * 16 + (lane & 15);
  int khalf = (lane >> 4) << 3;
  int k = (c << 5) + ((e < 8) ? (khalf + e) : (16 + khalf + (e - 8)));
  Bp[id] = (n < N) ? W[(size_t)k * N + n] : 0.f;
}

// ---------------------------------------------------------------------------
// Tiled GEMM: C[M,N] = epilogue(A[M,K] @ W[K,N] + bias (+res)).
// Block = 256 threads = 8 waves; wave tile 16x16; block tile 32x64.
// K multiple of 32; M multiple of 32; N guarded at store only (B is packed).
// fp32 -> (hi,lo) f16 split; acc fp32 via WMMA (3 wmma per k-step).
// FLAGS is a compile-time epilogue selector (0 / GF_GELU / GF_RES).
// ---------------------------------------------------------------------------
template <int FLAGS>
__global__ void __launch_bounds__(256)
k_gemm_wmma(const float* __restrict__ A, const float* __restrict__ Bp,
            const float* __restrict__ bias, const float* __restrict__ Res,
            float* __restrict__ C, int M, int K, int N, int nT16)
{
  const int lane   = threadIdx.x & 31;
  const int wave   = threadIdx.x >> 5;
  const int m0     = blockIdx.y * 32 + (wave >> 2) * 16;
  const int n0     = blockIdx.x * 64 + (wave & 3) * 16;
  const int mloc   = lane & 15;
  const int khalf  = (lane >> 4) << 3;          // 0 or 8 (K half per lane group)
  int arow = m0 + mloc; if (arow >= M) arow = M - 1;   // M mult of 32; safety clamp

  const int ntile = blockIdx.x * 4 + (wave & 3);
  const float* bbase = Bp + ((size_t)ntile * 32 + lane) * 16;
  const size_t bstep = (size_t)nT16 * 512;      // floats per k-chunk

  v8f acc = {0.f,0.f,0.f,0.f,0.f,0.f,0.f,0.f};

  for (int k0 = 0; k0 < K; k0 += 32) {
    v16h ah, al, bh, bl;
    const float* ap = A + (size_t)arow * K + k0 + khalf;
    const float* bp = bbase + (size_t)(k0 >> 5) * bstep;
    #pragma unroll
    for (int e = 0; e < 8; ++e) {
      float x0 = ap[e];
      float x1 = ap[16 + e];
      _Float16 h0 = (_Float16)x0; ah[e]   = h0; al[e]   = (_Float16)(x0 - (float)h0);
      _Float16 h1 = (_Float16)x1; ah[8+e] = h1; al[8+e] = (_Float16)(x1 - (float)h1);
    }
    #pragma unroll
    for (int e = 0; e < 8; ++e) {
      float y0 = bp[e];
      float y1 = bp[8 + e];
      _Float16 h0 = (_Float16)y0; bh[e]   = h0; bl[e]   = (_Float16)(y0 - (float)h0);
      _Float16 h1 = (_Float16)y1; bh[8+e] = h1; bl[8+e] = (_Float16)(y1 - (float)h1);
    }
    // acc += Al*Bh + Ah*Bl + Ah*Bh  (small terms first)
    acc = __builtin_amdgcn_wmma_f32_16x16x32_f16(false, al, false, bh, (short)0, acc, false, false);
    acc = __builtin_amdgcn_wmma_f32_16x16x32_f16(false, ah, false, bl, (short)0, acc, false, false);
    acc = __builtin_amdgcn_wmma_f32_16x16x32_f16(false, ah, false, bh, (short)0, acc, false, false);
  }

  const int nn = n0 + (lane & 15);
  if (nn < N) {
    const float bv = bias[nn];
    #pragma unroll
    for (int r = 0; r < 8; ++r) {
      const int mm = m0 + ((lane >> 4) << 3) + r;   // C/D layout: VGPR r -> M
      float v = acc[r] + bv;
      if (FLAGS & GF_GELU) v = gelu_exact(v);
      if (FLAGS & GF_RES)  v += Res[(size_t)mm * N + nn];
      C[(size_t)mm * N + nn] = v;
    }
  }
}

// ---------------------------------------------------------------------------
// RMSNorm over rows of width Dw (128/256). One wave per row; shfl_xor reduce.
// ---------------------------------------------------------------------------
__global__ void __launch_bounds__(256)
k_rmsnorm(const float* __restrict__ in, const float* __restrict__ w,
          float* __restrict__ out, long long rows, int Dw)
{
  const int lane = threadIdx.x & 31;
  const long long row = (long long)blockIdx.x * 8 + (threadIdx.x >> 5);
  if (row >= rows) return;
  const float* ip = in + row * Dw;
  float ss = 0.f;
  for (int j = lane; j < Dw; j += 32) { float v = ip[j]; ss += v * v; }
  #pragma unroll
  for (int m = 16; m; m >>= 1) ss += __shfl_xor(ss, m, 32);
  const float r = rsqrtf(ss / (float)Dw + 1e-6f);
  float* op = out + row * Dw;
  for (int j = lane; j < Dw; j += 32) op[j] = ip[j] * r * w[j];
}

// ---------------------------------------------------------------------------
// Embedding: h[n,t,v,d] = sum_c x[n,c,t,v]*ew[c,d] + eb[d] + pe[v,d]
// ---------------------------------------------------------------------------
__global__ void k_embed(const float* __restrict__ x, const float* __restrict__ ew,
                        const float* __restrict__ eb, const float* __restrict__ pe,
                        float* __restrict__ h, int Nn, int Tt)
{
  long long id = (long long)blockIdx.x * blockDim.x + threadIdx.x;
  long long total = (long long)Nn * Tt * 25 * 256;
  if (id >= total) return;
  int d = (int)(id & 255);
  long long rest = id >> 8;
  int v = (int)(rest % 25); rest /= 25;
  int t = (int)(rest % Tt);
  int n = (int)(rest / Tt);
  float a = eb[d] + pe[v * 256 + d];
  #pragma unroll
  for (int c = 0; c < 3; ++c)
    a += x[((size_t)(n * 3 + c) * Tt + t) * 25 + v] * ew[c * 256 + d];
  h[id] = a;
}

// ---------------------------------------------------------------------------
// Joint mixing: out[nt,u,d] = ep( sum_v in[nt,v,d] * w[v,u] + b[u] ), V=25.
// ---------------------------------------------------------------------------
__global__ void k_vmix(const float* __restrict__ in, const float* __restrict__ w,
                       const float* __restrict__ b, const float* __restrict__ res,
                       float* __restrict__ out, long long NT, int flags)
{
  long long id = (long long)blockIdx.x * blockDim.x + threadIdx.x;
  long long total = NT * 25 * 256;
  if (id >= total) return;
  int d = (int)(id & 255);
  long long rest = id >> 8;
  int u = (int)(rest % 25);
  long long nt = rest / 25;
  const float* ip = in + (size_t)nt * 25 * 256 + d;
  float a = 0.f;
  #pragma unroll
  for (int v = 0; v < 25; ++v) a += ip[(size_t)v * 256] * w[v * 25 + u];
  a += b[u];
  if (flags & GF_GELU) a = gelu_exact(a);
  if (flags & GF_RES)  a += res[id];
  out[id] = a;
}

// (N,T,V,D) -> (N*V,T,D)
__global__ void k_h2t(const float* __restrict__ h, float* __restrict__ ht, int Nn, int Tt)
{
  long long id = (long long)blockIdx.x * blockDim.x + threadIdx.x;
  long long total = (long long)Nn * Tt * 25 * 256;
  if (id >= total) return;
  int d = (int)(id & 255);
  long long rest = id >> 8;
  int v = (int)(rest % 25); rest /= 25;
  int t = (int)(rest % Tt);
  int n = (int)(rest / Tt);
  ht[(((size_t)(n * 25 + v) * Tt) + t) * 256 + d] = h[id];
}

// (N*V,T,D) -> (N,T,V,D)
__global__ void k_t2h(const float* __restrict__ ht, float* __restrict__ h, int Nn, int Tt)
{
  long long id = (long long)blockIdx.x * blockDim.x + threadIdx.x;
  long long total = (long long)Nn * Tt * 25 * 256;
  if (id >= total) return;
  int d = (int)(id & 255);
  long long rest = id >> 8;
  int t = (int)(rest % Tt); rest /= Tt;
  int v = (int)(rest % 25);
  int n = (int)(rest / 25);
  h[(((size_t)(n * Tt + t) * 25) + v) * 256 + d] = ht[id];
}

// pad + roll(-shift): xi[r,tp,d] = (src<Tt)? nrm[r,src,d] : 0 ; src=(tp+shift)%Tp
__global__ void k_buildxi(const float* __restrict__ nrm, float* __restrict__ xi,
                          int R, int Tt, int Tp, int shift)
{
  long long id = (long long)blockIdx.x * blockDim.x + threadIdx.x;
  long long total = (long long)R * Tp * 128;
  if (id >= total) return;
  int d = (int)(id & 127);
  long long rest = id >> 7;
  int tp = (int)(rest % Tp);
  int r  = (int)(rest / Tp);
  int src = (tp + shift) % Tp;
  xi[id] = (src < Tt) ? nrm[((size_t)r * Tt + src) * 128 + d] : 0.f;
}

// roll(+shift) + unpad: out[r,t,d] = in[r,(t-shift mod Tp),d], t<Tt
__global__ void k_gather(const float* __restrict__ in, float* __restrict__ out,
                         int R, int Tt, int Tp, int shift)
{
  long long id = (long long)blockIdx.x * blockDim.x + threadIdx.x;
  long long total = (long long)R * Tt * 128;
  if (id >= total) return;
  int d = (int)(id & 127);
  long long rest = id >> 7;
  int t = (int)(rest % Tt);
  int r = (int)(rest / Tt);
  int st = ((t - shift) % Tp + Tp) % Tp;
  out[id] = in[((size_t)r * Tp + st) * 128 + d];
}

// ---------------------------------------------------------------------------
// Windowed attention, W=20, H=4, hd=32, DIN=128. One thread per (win,head,row).
// QKV row layout: [q(h*32+e) | k | v] = 384 floats. Swin shift mask inline.
// ---------------------------------------------------------------------------
__global__ void k_win_attn(const float* __restrict__ QKV, float* __restrict__ O,
                           const float* __restrict__ rpb, int nW, int Tp, int shift,
                           long long total)
{
  long long id = (long long)blockIdx.x * blockDim.x + threadIdx.x;
  if (id >= total) return;
  int i   = (int)(id % 20);
  int h   = (int)((id / 20) % 4);
  int wdw = (int)((id / 80) % nW);
  int r   = (int)(id / (80LL * nW));
  const int base = r * Tp + wdw * 20;
  const float scale = 0.17677669529663687f;  // 1/sqrt(32)
  const int ti = wdw * 20 + i;
  const int segi = (shift > 0) ? ((ti < Tp - 20) ? 0 : ((ti < Tp - shift) ? 1 : 2)) : 0;

  float s[20];
  const float* q = QKV + (size_t)(base + i) * 384 + h * 32;
  for (int j = 0; j < 20; ++j) {
    const float* kp = QKV + (size_t)(base + j) * 384 + 128 + h * 32;
    float dsum = 0.f;
    #pragma unroll
    for (int e = 0; e < 32; ++e) dsum += q[e] * kp[e];
    float a = dsum * scale + rpb[(i - j + 19) * 4 + h];
    if (shift > 0) {
      int tj = wdw * 20 + j;
      int segj = (tj < Tp - 20) ? 0 : ((tj < Tp - shift) ? 1 : 2);
      if (segj != segi) a += -100.f;
    }
    s[j] = a;
  }
  float m = s[0];
  for (int j = 1; j < 20; ++j) m = fmaxf(m, s[j]);
  float Z = 0.f;
  for (int j = 0; j < 20; ++j) { s[j] = expf(s[j] - m); Z += s[j]; }
  const float invZ = 1.f / Z;
  float* op = O + (size_t)(base + i) * 128 + h * 32;
  for (int e = 0; e < 32; ++e) {
    float a = 0.f;
    for (int j = 0; j < 20; ++j)
      a += s[j] * QKV[(size_t)(base + j) * 384 + 256 + h * 32 + e];
    op[e] = a * invZ;
  }
}

// im2col for conv1d k=3, stride 2, pad 1: col[(r,t2), i*3+k] = ht[r, 2*t2-1+k, i]
__global__ void k_im2col(const float* __restrict__ ht, float* __restrict__ col,
                         int R, int Tt, int T2)
{
  long long id = (long long)blockIdx.x * blockDim.x + threadIdx.x;
  long long total = (long long)R * T2 * 768;
  if (id >= total) return;
  int c  = (int)(id % 768);
  long long rest = id / 768;
  int t2 = (int)(rest % T2);
  int r  = (int)(rest / T2);
  int i = c / 3, k = c % 3;
  int ts = 2 * t2 - 1 + k;
  col[id] = (ts >= 0 && ts < Tt) ? ht[((size_t)r * Tt + ts) * 256 + i] : 0.f;
}

// ds_w (O,I,K) -> Wm[(i*3+k), o]
__global__ void k_wtrans(const float* __restrict__ dsw, float* __restrict__ Wm)
{
  int id = blockIdx.x * blockDim.x + threadIdx.x;   // 768*256
  if (id >= 768 * 256) return;
  int o = id & 255;
  int c = id >> 8;            // i*3+k
  Wm[id] = dsw[(size_t)o * 768 + c];
}

// mean over V: ph[n,t,d] = mean_v ht[(n*25+v), t, d]
__global__ void k_meanV(const float* __restrict__ ht, float* __restrict__ ph, int Nn, int T2)
{
  long long id = (long long)blockIdx.x * blockDim.x + threadIdx.x;
  long long total = (long long)Nn * T2 * 256;
  if (id >= total) return;
  int d = (int)(id & 255);
  long long rest = id >> 8;
  int t = (int)(rest % T2);
  int n = (int)(rest / T2);
  float a = 0.f;
  #pragma unroll
  for (int v = 0; v < 25; ++v)
    a += ht[(((size_t)(n * 25 + v)) * T2 + t) * 256 + d];
  ph[id] = a * 0.04f;
}

// per-(n,t): softmax(60) -> entropy -> w = 1 + 1/(1+ent); normalize over T2.
__global__ void k_timeweights(const float* __restrict__ logits, float* __restrict__ wts, int T2)
{
  __shared__ float red[128];
  const int n = blockIdx.x, t = threadIdx.x;
  const float* lp = logits + (size_t)(n * T2 + t) * 60;
  float m = -1e30f;
  for (int c = 0; c < 60; ++c) m = fmaxf(m, lp[c]);
  float Z = 0.f;
  for (int c = 0; c < 60; ++c) Z += expf(lp[c] - m);
  float ent = 0.f;
  for (int c = 0; c < 60; ++c) { float p = expf(lp[c] - m) / Z; ent -= p * logf(p + 1e-8f); }
  const float wraw = 1.f + 1.f / (1.f + ent);
  red[t] = wraw;
  __syncthreads();
  for (int s = 64; s; s >>= 1) { if (t < s) red[t] += red[t + s]; __syncthreads(); }
  wts[n * T2 + t] = wraw / (red[0] + 1e-8f);
}

// pooled[n,d] = sum_t wts[n,t] * ph[n,t,d]
__global__ void k_wpool(const float* __restrict__ ph, const float* __restrict__ wts,
                        float* __restrict__ pooled, int T2)
{
  const int n = blockIdx.x, d = threadIdx.x;
  float a = 0.f;
  for (int t = 0; t < T2; ++t)
    a += wts[n * T2 + t] * ph[((size_t)(n * T2 + t)) * 256 + d];
  pooled[n * 256 + d] = a;
}

// ============================ host orchestration ============================

static inline int eblk(long long n) { return (int)((n + 255) / 256); }

// Packs W into pk (fragment order), then runs the WMMA GEMM. All launches on
// one stream, so the single pk slab is safely reused across sequential GEMMs.
static void gemm(const float* A, const float* Wt, const float* b, const float* Res,
                 float* C, int M, int K, int N, int flags, float* pk, hipStream_t s)
{
  const int nb64 = (N + 63) / 64;
  const int nT16 = nb64 * 4;                    // padded 16-col tiles
  const int ptotal = (K >> 5) * nT16 * 512;
  k_packB<<<(ptotal + 255) / 256, 256, 0, s>>>(Wt, pk, K, N, nT16);
  dim3 g(nb64, M / 32);
  if (flags == GF_GELU)
    k_gemm_wmma<GF_GELU><<<g, 256, 0, s>>>(A, pk, b, Res, C, M, K, N, nT16);
  else if (flags == GF_RES)
    k_gemm_wmma<GF_RES><<<g, 256, 0, s>>>(A, pk, b, Res, C, M, K, N, nT16);
  else
    k_gemm_wmma<0><<<g, 256, 0, s>>>(A, pk, b, Res, C, M, K, N, nT16);
}

static void rms(const float* in, const float* w, float* out, long long rows, int Dw, hipStream_t s)
{
  k_rmsnorm<<<(int)((rows + 7) / 8), 256, 0, s>>>(in, w, out, rows, Dw);
}

// spatial block params at base index: n1,t1w,t1b,t2w,t2b,n2,c1w,c1b,c2w,c2b
static void run_spatial(float* X, float* Dd, float* Ab, const float* const* P, int bs,
                        long long NT, float* pk, hipStream_t s)
{
  long long M = NT * 25;
  rms(X, P[bs + 0], Dd, M, 256, s);
  k_vmix<<<eblk(M * 256), 256, 0, s>>>(Dd, P[bs + 1], P[bs + 2], nullptr, Ab, NT, GF_GELU);
  k_vmix<<<eblk(M * 256), 256, 0, s>>>(Ab, P[bs + 3], P[bs + 4], X, X, NT, GF_RES);
  rms(X, P[bs + 5], Dd, M, 256, s);
  gemm(Dd, P[bs + 6], P[bs + 7], nullptr, Ab, (int)M, 256, 512, GF_GELU, pk, s);
  gemm(Ab, P[bs + 8], P[bs + 9], X, X, (int)M, 512, 256, GF_RES, pk, s);
}

// swin params at base: pdw,pdb,n1,rpb,qkvw,qkvb,projw,projb,puw,pub,n2,f1w,f1b,f2w,f2b
// Result lands back in X. Tm is a scratch slab >= R*Tt*256 floats.
static void run_swin(float* X, float* Tm, float* Dd, float* E, float* Ab,
                     const float* const* P, int bs, int R, int Tt, int shift,
                     float* pk, hipStream_t s)
{
  const int M   = R * Tt;
  const int pad = (20 - Tt % 20) % 20;
  const int Tp  = Tt + pad;
  const int nW  = Tp / 20;
  const int M2  = R * Tp;

  gemm(X, P[bs + 0], P[bs + 1], nullptr, Dd, M, 256, 128, 0, pk, s);    // pd
  rms(Dd, P[bs + 2], Dd, M, 128, s);                                    // n1 (in place)
  k_buildxi<<<eblk((long long)M2 * 128), 256, 0, s>>>(Dd, E, R, Tt, Tp, shift);
  gemm(E, P[bs + 4], P[bs + 5], nullptr, Ab, M2, 128, 384, 0, pk, s);   // qkv
  k_win_attn<<<eblk((long long)R * nW * 80), 256, 0, s>>>(Ab, Tm, P[bs + 3], nW, Tp, shift,
                                                          (long long)R * nW * 80);
  gemm(Tm, P[bs + 6], P[bs + 7], nullptr, E, M2, 128, 128, 0, pk, s);   // proj
  k_gather<<<eblk((long long)M * 128), 256, 0, s>>>(E, Dd, R, Tt, Tp, shift);
  gemm(Dd, P[bs + 8], P[bs + 9], X, Tm, M, 128, 256, GF_RES, pk, s);    // pu + residual -> Tm
  rms(Tm, P[bs + 10], Dd, M, 256, s);                                   // n2
  gemm(Dd, P[bs + 11], P[bs + 12], nullptr, Ab, M, 256, 512, GF_GELU, pk, s); // f1
  gemm(Ab, P[bs + 13], P[bs + 14], Tm, X, M, 512, 256, GF_RES, pk, s);  // f2 -> X
}

extern "C" void kernel_launch(void* const* d_in, const int* in_sizes, int n_in,
                              void* d_out, int out_size, void* d_ws, size_t ws_size,
                              hipStream_t stream)
{
  (void)in_sizes; (void)out_size;
  if (n_in < 87) return;

  const float* P[96];
  for (int i = 0; i < n_in && i < 96; ++i) P[i] = (const float*)d_in[i];
  // d_in order (setup_inputs insertion order):
  // 0:x 1:emb.w 2:emb.b 3:pe  4..13:sp1  14..23:sp2  24..33:sp3
  // 34..48:tp1 49..63:tp2 64..78:tp3  79:ds_w 80:ds_b 81:ds_n
  // 82:pool.w 83:pool.b 84:head_n 85:head.w 86:head.b

  // workspace slabs (floats)
  float* ws = (float*)d_ws;
  const size_t offA  = 0;                         // 104,857,600 (hidden/qkv/im2col)
  const size_t offB  = offA + 104857600ull;       //  52,428,800
  const size_t offC  = offB + 52428800ull;        //  52,428,800
  const size_t offD  = offC + 52428800ull;        //  52,428,800
  const size_t offE  = offD + 52428800ull;        //  26,624,000 (padded xi)
  const size_t offW  = offE + 26624000ull;        //     196,608 (conv weight, GEMM form)
  const size_t offPK = offW + 196608ull;          //     196,608 (packed-B scratch)
  const size_t offPH = offPK + 196608ull;         //   1,048,576 (pooled_h)
  const size_t offLG = offPH + 1048576ull;        //     245,760 (logits)
  const size_t offWT = offLG + 245760ull;         //       4,096 (wts)
  const size_t offP1 = offWT + 4096ull;           //       8,192 (pooled)
  const size_t offP2 = offP1 + 8192ull;           //       8,192 (pooled normed)
  const size_t needF = offP2 + 8192ull;
  if (ws_size < needF * sizeof(float)) return;

  float* Ab = ws + offA;  float* Bb = ws + offB;  float* Cb = ws + offC;
  float* Db = ws + offD;  float* Eb = ws + offE;  float* Wm = ws + offW;
  float* pk = ws + offPK;
  float* ph = ws + offPH; float* lg = ws + offLG; float* wt = ws + offWT;
  float* p1 = ws + offP1; float* p2 = ws + offP2;

  const int Nn = 32, Tt = 256, T2 = 128;
  const long long tok0 = (long long)Nn * Tt * 25;   // 204800

  // ---- stem ----
  k_embed<<<eblk(tok0 * 256), 256, 0, stream>>>(P[0], P[1], P[2], P[3], Bb, Nn, Tt);
  run_spatial(Bb, Db, Ab, P, 4, (long long)Nn * Tt, pk, stream);        // sp1
  k_h2t<<<eblk(tok0 * 256), 256, 0, stream>>>(Bb, Cb, Nn, Tt);
  run_swin(Cb, Bb, Db, Eb, Ab, P, 34, 800, Tt, 0, pk, stream);          // tp1 -> Cb

  // ---- downsample (conv k=3 s=2 p=1 as im2col GEMM) ----
  k_wtrans<<<(768 * 256 + 255) / 256, 256, 0, stream>>>(P[79], Wm);
  k_im2col<<<eblk(800LL * T2 * 768), 256, 0, stream>>>(Cb, Ab, 800, Tt, T2);
  gemm(Ab, Wm, P[80], nullptr, Bb, 800 * T2, 768, 256, 0, pk, stream);
  rms(Bb, P[81], Bb, (long long)800 * T2, 256, stream);                 // ht2 in Bb

  const long long tok1 = (long long)Nn * T2 * 25;   // 102400
  k_t2h<<<eblk(tok1 * 256), 256, 0, stream>>>(Bb, Cb, Nn, T2);
  run_spatial(Cb, Db, Ab, P, 14, (long long)Nn * T2, pk, stream);       // sp2
  k_h2t<<<eblk(tok1 * 256), 256, 0, stream>>>(Cb, Bb, Nn, T2);
  run_swin(Bb, Cb, Db, Eb, Ab, P, 49, 800, T2, 10, pk, stream);         // tp2 (shift W/2)
  k_t2h<<<eblk(tok1 * 256), 256, 0, stream>>>(Bb, Cb, Nn, T2);
  run_spatial(Cb, Db, Ab, P, 24, (long long)Nn * T2, pk, stream);       // sp3
  k_h2t<<<eblk(tok1 * 256), 256, 0, stream>>>(Cb, Bb, Nn, T2);
  run_swin(Bb, Cb, Db, Eb, Ab, P, 64, 800, T2, 0, pk, stream);          // tp3 -> Bb

  // ---- head: mean over V, entropy-weighted time pooling, classifier ----
  k_meanV<<<eblk((long long)Nn * T2 * 256), 256, 0, stream>>>(Bb, ph, Nn, T2);
  gemm(ph, P[82], P[83], nullptr, lg, Nn * T2, 256, 60, 0, pk, stream); // pool logits
  k_timeweights<<<Nn, T2, 0, stream>>>(lg, wt, T2);
  k_wpool<<<Nn, 256, 0, stream>>>(ph, wt, p1, T2);
  rms(p1, P[84], p2, Nn, 256, stream);
  gemm(p2, P[85], P[86], nullptr, (float*)d_out, 32, 256, 60, 0, pk, stream); // final (32x60)
}
